// RWKV7Attention_55155970015404
// MI455X (gfx1250) — compile-verified, matrix operand order checked by reading, not run on hardware
//
#include <hip/hip_runtime.h>
#include <hip/hip_bf16.h>

#define Bsz   2
#define Tlen  1024
#define Cdim  2048
#define Hn    32
#define Nn    64
#define Mrows (Bsz * Tlen)

typedef __bf16 bf16_t;
typedef bf16_t v16bf __attribute__((ext_vector_type(16)));
typedef float  v8f   __attribute__((ext_vector_type(8)));

#if defined(__has_builtin)
#if __has_builtin(__builtin_amdgcn_global_load_async_to_lds_b32)
#define HAVE_ASYNC_LDS 1
#endif
#endif

// ---------------------------------------------------------------- helpers

// native f32 -> bf16 (should select v_cvt_*bf16* on gfx1250)
__device__ inline bf16_t f2bf(float f) { return (bf16_t)f; }

__device__ inline float sigmoidf_(float x) { return 1.0f / (1.0f + __expf(-x)); }

__device__ inline void load16f(const float* __restrict__ p, float* d) {
  const float4* p4 = reinterpret_cast<const float4*>(p);
  float4 a = p4[0], b = p4[1], c = p4[2], e = p4[3];
  d[0]=a.x; d[1]=a.y; d[2]=a.z;  d[3]=a.w;
  d[4]=b.x; d[5]=b.y; d[6]=b.z;  d[7]=b.w;
  d[8]=c.x; d[9]=c.y; d[10]=c.z; d[11]=c.w;
  d[12]=e.x; d[13]=e.y; d[14]=e.z; d[15]=e.w;
}

// ------------------------------------------- weight transpose + bf16 convert
// in: fp32 row-major [K, N]  ->  out: bf16 row-major [N, K]  (B operand layout)
// LDS-tiled 32x32 so both global read and global write are coalesced.
// Requires K % 32 == 0 and N % 32 == 0 (true for all matrices here).
__global__ void tcvt_kernel(const float* __restrict__ in, bf16_t* __restrict__ out,
                            int K, int N) {
  __shared__ bf16_t tile[32][33];
  int nb = blockIdx.x * 32;
  int kb = blockIdx.y * 32;
  int tx = threadIdx.x;          // 0..31
  int ty = threadIdx.y;          // 0..7
#pragma unroll
  for (int i = ty; i < 32; i += 8) {
    tile[i][tx] = f2bf(in[(size_t)(kb + i) * N + (nb + tx)]);
  }
  __syncthreads();
#pragma unroll
  for (int i = ty; i < 32; i += 8) {
    out[(size_t)(nb + i) * K + (kb + tx)] = tile[tx][i];
  }
}

// ---------------------------------------------------------- GEMM epilogue
// mode: 0 = f32 store, 1 = bf16 store, 2 = tanh->bf16, 3 = sigmoid->bf16
__device__ inline void epilogue_store(const v8f acc[2][2], int m0, int n0,
                                      int lrow, int half, int N,
                                      float* outF, bf16_t* outB, int mode) {
#pragma unroll
  for (int mi = 0; mi < 2; ++mi)
#pragma unroll
    for (int ni = 0; ni < 2; ++ni) {
      int n = n0 + ni * 16 + lrow;
#pragma unroll
      for (int vi = 0; vi < 8; ++vi) {
        int m = m0 + mi * 16 + half * 8 + vi;
        float val = acc[mi][ni][vi];
        if (mode == 0) {
          outF[(size_t)m * N + n] = val;
        } else {
          if (mode == 2)      val = tanhf(val);
          else if (mode == 3) val = sigmoidf_(val);
          outB[(size_t)m * N + n] = f2bf(val);
        }
      }
    }
}

// -------------------------------------- GEMM with fused token-shift mix (A)
// A(m,k) = x(m,k) + (xprev(m,k) - x(m,k)) * coef(k), converted to bf16.
// B operand: bf16 [N, K] (pre-transposed).  K == Cdim.
__global__ void gemm_mix_kernel(const float* __restrict__ X,
                                const float* __restrict__ coef,
                                const bf16_t* __restrict__ BT,
                                int N, int jobs,
                                float* outF, bf16_t* outB, int mode) {
  const int K = Cdim;
  int wave = threadIdx.x >> 5;
  int job  = blockIdx.x * (blockDim.x >> 5) + wave;
  if (job >= jobs) return;
  int jobs_n = N >> 5;
  int mt = job / jobs_n, nt = job - mt * jobs_n;
  int m0 = mt * 32, n0 = nt * 32;
  int lane = threadIdx.x & 31;
  int lrow = lane & 15, half = lane >> 4;

  int  mrow0 = m0 + lrow;                 // rows for A frag 0 (M 0..15)
  int  mrow1 = m0 + 16 + lrow;            // rows for A frag 1 (M 16..31)
  bool hasprev0 = (mrow0 % Tlen) != 0;    // frag1 rows always have t>=16
  const float*  xr0 = X + (size_t)mrow0 * K;
  const float*  xr1 = X + (size_t)mrow1 * K;
  const bf16_t* bp0 = BT + (size_t)(n0 + lrow) * K;
  const bf16_t* bp1 = BT + (size_t)(n0 + 16 + lrow) * K;

  v8f acc[2][2] = {};
  for (int k0 = 0; k0 < K; k0 += 32) {
    int kk = k0 + half * 16;
    if (k0 + 32 < K) {
      __builtin_prefetch(bp0 + kk + 32, 0, 1);
      __builtin_prefetch(bp1 + kk + 32, 0, 1);
    }
    float cf[16]; load16f(coef + kk, cf);
    float x0[16], x1[16], p0[16], p1[16];
    load16f(xr0 + kk, x0);
    load16f(xr1 + kk, x1);
    if (hasprev0) {
      load16f(xr0 + kk - K, p0);
    } else {
#pragma unroll
      for (int i = 0; i < 16; ++i) p0[i] = 0.f;
    }
    load16f(xr1 + kk - K, p1);
    v16bf a0, a1;
#pragma unroll
    for (int i = 0; i < 16; ++i) {
      a0[i] = f2bf(x0[i] + (p0[i] - x0[i]) * cf[i]);
      a1[i] = f2bf(x1[i] + (p1[i] - x1[i]) * cf[i]);
    }
    v16bf bf0 = *reinterpret_cast<const v16bf*>(bp0 + kk);
    v16bf bf1 = *reinterpret_cast<const v16bf*>(bp1 + kk);
    acc[0][0] = __builtin_amdgcn_wmma_f32_16x16x32_bf16(false, a0, false, bf0, (short)0, acc[0][0], false, false);
    acc[0][1] = __builtin_amdgcn_wmma_f32_16x16x32_bf16(false, a0, false, bf1, (short)0, acc[0][1], false, false);
    acc[1][0] = __builtin_amdgcn_wmma_f32_16x16x32_bf16(false, a1, false, bf0, (short)0, acc[1][0], false, false);
    acc[1][1] = __builtin_amdgcn_wmma_f32_16x16x32_bf16(false, a1, false, bf1, (short)0, acc[1][1], false, false);
  }
  epilogue_store(acc, m0, n0, lrow, half, N, outF, outB, mode);
}

// ------------------------------------------------ plain bf16 A-operand GEMM
__global__ void gemm_bf16_kernel(const bf16_t* __restrict__ A,
                                 const bf16_t* __restrict__ BT,
                                 int N, int K, int jobs,
                                 float* outF, bf16_t* outB, int mode) {
  int wave = threadIdx.x >> 5;
  int job  = blockIdx.x * (blockDim.x >> 5) + wave;
  if (job >= jobs) return;
  int jobs_n = N >> 5;
  int mt = job / jobs_n, nt = job - mt * jobs_n;
  int m0 = mt * 32, n0 = nt * 32;
  int lane = threadIdx.x & 31;
  int lrow = lane & 15, half = lane >> 4;

  const bf16_t* ap0 = A + (size_t)(m0 + lrow) * K;
  const bf16_t* ap1 = A + (size_t)(m0 + 16 + lrow) * K;
  const bf16_t* bp0 = BT + (size_t)(n0 + lrow) * K;
  const bf16_t* bp1 = BT + (size_t)(n0 + 16 + lrow) * K;

  v8f acc[2][2] = {};
  for (int k0 = 0; k0 < K; k0 += 32) {
    int kk = k0 + half * 16;
    if (k0 + 32 < K) {
      __builtin_prefetch(bp0 + kk + 32, 0, 1);
      __builtin_prefetch(bp1 + kk + 32, 0, 1);
    }
    v16bf a0 = *reinterpret_cast<const v16bf*>(ap0 + kk);
    v16bf a1 = *reinterpret_cast<const v16bf*>(ap1 + kk);
    v16bf b0 = *reinterpret_cast<const v16bf*>(bp0 + kk);
    v16bf b1 = *reinterpret_cast<const v16bf*>(bp1 + kk);
    acc[0][0] = __builtin_amdgcn_wmma_f32_16x16x32_bf16(false, a0, false, b0, (short)0, acc[0][0], false, false);
    acc[0][1] = __builtin_amdgcn_wmma_f32_16x16x32_bf16(false, a0, false, b1, (short)0, acc[0][1], false, false);
    acc[1][0] = __builtin_amdgcn_wmma_f32_16x16x32_bf16(false, a1, false, b0, (short)0, acc[1][0], false, false);
    acc[1][1] = __builtin_amdgcn_wmma_f32_16x16x32_bf16(false, a1, false, b1, (short)0, acc[1][1], false, false);
  }
  epilogue_store(acc, m0, n0, lrow, half, N, outF, outB, mode);
}

// --------------------------------------------- per-(b,t,h) elementwise prep
// one 64-thread block per (b,t,h); in-place over the LoRA/raw buffers.
__global__ void post_kernel(const float* kraw, const float* vraw,
                            const float* vfirst,
                            const float* wl, const float* au, const float* vu,
                            const float* k_k, const float* k_a,
                            const float* a0v, const float* w0v, const float* v0v,
                            float* wout, float* kout, float* vout,
                            float* aaout, float* bbout) {
  int blk = blockIdx.x;
  int h = blk & (Hn - 1);
  int m = blk >> 5;               // row in [0, Mrows)
  int n = threadIdx.x;            // head channel
  int c = h * Nn + n;
  size_t idx = (size_t)m * Cdim + c;

  float kr  = kraw[idx];
  float kkv = kr * k_k[c];

  __shared__ float red[Nn];
  red[n] = kkv * kkv;
  __syncthreads();
  float ss = 0.f;
#pragma unroll
  for (int j = 0; j < Nn; ++j) ss += red[j];
  float norm = sqrtf(ss);
  float kkn  = kkv / fmaxf(norm, 1e-12f);

  float a  = sigmoidf_(a0v[c] + au[idx]);
  float x  = -(w0v[c] + wl[idx]);
  float sp = (x > 20.f) ? x : log1pf(__expf(x));    // softplus(-(w0+wl))
  float w  = __expf(-__expf(-0.5f - sp));
  float kf = kr * (1.0f + (a - 1.0f) * k_a[c]);
  float vr = vraw[idx];
  float vf = vr + (vfirst[idx] - vr) * sigmoidf_(v0v[c] + vu[idx]);

  wout[idx]  = w;
  kout[idx]  = kf;
  vout[idx]  = vf;
  aaout[idx] = -kkn;
  bbout[idx] = kkn * a;
}

// ------------------------------------------------------- WKV7 recurrent scan
// grid = B*H blocks of 64 threads; thread tid owns state row S[v=tid][k=0..63].
// Per-timestep vectors staged to LDS via async load-to-LDS when available.
__global__ void wkv_scan_kernel(const float* __restrict__ r,
                                const float* __restrict__ w,
                                const float* __restrict__ k,
                                const float* __restrict__ v,
                                const float* __restrict__ aa,
                                const float* __restrict__ bb,
                                float* __restrict__ y) {
  int bh  = blockIdx.x;
  int b   = bh / Hn, h = bh - b * Hn;
  int tid = threadIdx.x;

  float S[Nn];
#pragma unroll
  for (int j = 0; j < Nn; ++j) S[j] = 0.f;

  __shared__ float lr[Nn], lw[Nn], lk[Nn], lv[Nn], la[Nn], lb[Nn];
  size_t base = ((size_t)b * Tlen) * Cdim + (size_t)h * Nn + tid;

  for (int t = 0; t < Tlen; ++t) {
    size_t idx = base + (size_t)t * Cdim;
#ifdef HAVE_ASYNC_LDS
    typedef __attribute__((address_space(1))) int gint_t;   // global AS
    typedef __attribute__((address_space(3))) int lint_t;   // LDS AS
    __builtin_amdgcn_global_load_async_to_lds_b32(
        (gint_t*)(const_cast<float*>(r)  + idx), (lint_t*)&lr[tid], 0, 0);
    __builtin_amdgcn_global_load_async_to_lds_b32(
        (gint_t*)(const_cast<float*>(w)  + idx), (lint_t*)&lw[tid], 0, 0);
    __builtin_amdgcn_global_load_async_to_lds_b32(
        (gint_t*)(const_cast<float*>(k)  + idx), (lint_t*)&lk[tid], 0, 0);
    __builtin_amdgcn_global_load_async_to_lds_b32(
        (gint_t*)(const_cast<float*>(v)  + idx), (lint_t*)&lv[tid], 0, 0);
    __builtin_amdgcn_global_load_async_to_lds_b32(
        (gint_t*)(const_cast<float*>(aa) + idx), (lint_t*)&la[tid], 0, 0);
    __builtin_amdgcn_global_load_async_to_lds_b32(
        (gint_t*)(const_cast<float*>(bb) + idx), (lint_t*)&lb[tid], 0, 0);
#if __has_builtin(__builtin_amdgcn_s_wait_asynccnt)
    __builtin_amdgcn_s_wait_asynccnt(0);
#else
    asm volatile("s_wait_asynccnt 0x0" ::: "memory");
#endif
#else
    lr[tid] = r[idx];  lw[tid] = w[idx];  lk[tid] = k[idx];
    lv[tid] = v[idx];  la[tid] = aa[idx]; lb[tid] = bb[idx];
#endif
    __syncthreads();

    float sa = 0.f;
#pragma unroll
    for (int j = 0; j < Nn; ++j) sa += S[j] * la[j];

    float vt = lv[tid];
    float yv = 0.f;
#pragma unroll
    for (int j = 0; j < Nn; ++j) {
      float s = S[j] * lw[j] + sa * lb[j] + vt * lk[j];
      S[j] = s;
      yv += s * lr[j];
    }
    y[idx] = yv;
    __syncthreads();
  }
}

// ------------------------------------ GroupNorm + r*k*r_k bonus + gate -> bf16
__global__ void finalize_kernel(const float* y, const float* r, const float* k,
                                const float* v, const float* g,
                                const float* r_k, const float* gn_w,
                                const float* gn_b, bf16_t* ypre) {
  int blk = blockIdx.x;
  int h = blk & (Hn - 1);
  int m = blk >> 5;
  int n = threadIdx.x;
  int c = h * Nn + n;
  size_t idx = (size_t)m * Cdim + c;

  float yv = y[idx];
  __shared__ float red[Nn];

  red[n] = yv;
  __syncthreads();
  float mu = 0.f;
#pragma unroll
  for (int j = 0; j < Nn; ++j) mu += red[j];
  mu *= (1.0f / Nn);
  __syncthreads();

  float d = yv - mu;
  red[n] = d * d;
  __syncthreads();
  float var = 0.f;
#pragma unroll
  for (int j = 0; j < Nn; ++j) var += red[j];
  var *= (1.0f / Nn);
  float xn = d * rsqrtf(var + 6.4e-4f);       // eps = N * 1e-5
  float y1 = xn * gn_w[c] + gn_b[c];
  __syncthreads();

  float rv = r[idx], kv = k[idx];
  red[n] = rv * kv * r_k[h * Nn + n];
  __syncthreads();
  float s = 0.f;
#pragma unroll
  for (int j = 0; j < Nn; ++j) s += red[j];

  float y2 = y1 + s * v[idx];
  ypre[idx] = f2bf(y2 * g[idx]);
}

// ================================================================= launcher

extern "C" void kernel_launch(void* const* d_in, const int* in_sizes, int n_in,
                              void* d_out, int out_size, void* d_ws, size_t ws_size,
                              hipStream_t stream) {
  (void)in_sizes; (void)n_in; (void)out_size; (void)ws_size;

  const float* hidden  = (const float*)d_in[0];
  const float* v_first = (const float*)d_in[1];
  const float* x_r = (const float*)d_in[2];
  const float* x_w = (const float*)d_in[3];
  const float* x_k = (const float*)d_in[4];
  const float* x_v = (const float*)d_in[5];
  const float* x_a = (const float*)d_in[6];
  const float* x_g = (const float*)d_in[7];
  const float* w0  = (const float*)d_in[8];
  const float* w1  = (const float*)d_in[9];
  const float* w2  = (const float*)d_in[10];
  const float* a0  = (const float*)d_in[11];
  const float* a1  = (const float*)d_in[12];
  const float* a2  = (const float*)d_in[13];
  const float* v0  = (const float*)d_in[14];
  const float* v1  = (const float*)d_in[15];
  const float* v2  = (const float*)d_in[16];
  const float* g1  = (const float*)d_in[17];
  const float* g2  = (const float*)d_in[18];
  const float* k_k = (const float*)d_in[19];
  const float* k_a = (const float*)d_in[20];
  const float* r_k = (const float*)d_in[21];
  const float* W_r = (const float*)d_in[22];
  const float* W_k = (const float*)d_in[23];
  const float* W_v = (const float*)d_in[24];
  const float* W_o = (const float*)d_in[25];
  const float* gn_w = (const float*)d_in[26];
  const float* gn_b = (const float*)d_in[27];

  char* base = (char*)d_ws;
  size_t off = 0;
  auto alloc = [&](size_t bytes) -> void* {
    off = (off + 255) & ~(size_t)255;
    void* p = base + off;
    off += bytes;
    return p;
  };

  const size_t MC = (size_t)Mrows * Cdim;

  // bf16 transposed weights (B operands)
  bf16_t* WrT = (bf16_t*)alloc(MC * 2);
  bf16_t* WkT = (bf16_t*)alloc(MC * 2);
  bf16_t* WvT = (bf16_t*)alloc(MC * 2);
  bf16_t* WoT = (bf16_t*)alloc(MC * 2);
  bf16_t* w1T = (bf16_t*)alloc((size_t)96  * Cdim * 2);
  bf16_t* w2T = (bf16_t*)alloc((size_t)Cdim * 96  * 2);
  bf16_t* a1T = (bf16_t*)alloc((size_t)96  * Cdim * 2);
  bf16_t* a2T = (bf16_t*)alloc((size_t)Cdim * 96  * 2);
  bf16_t* v1T = (bf16_t*)alloc((size_t)64  * Cdim * 2);
  bf16_t* v2T = (bf16_t*)alloc((size_t)Cdim * 64  * 2);
  bf16_t* g1T = (bf16_t*)alloc((size_t)256 * Cdim * 2);
  bf16_t* g2T = (bf16_t*)alloc((size_t)Cdim * 256 * 2);
  // bf16 LoRA down activations + pre-output
  bf16_t* wdown = (bf16_t*)alloc((size_t)Mrows * 96  * 2);
  bf16_t* adown = (bf16_t*)alloc((size_t)Mrows * 96  * 2);
  bf16_t* vdown = (bf16_t*)alloc((size_t)Mrows * 64  * 2);
  bf16_t* gdown = (bf16_t*)alloc((size_t)Mrows * 256 * 2);
  bf16_t* ypre  = (bf16_t*)alloc(MC * 2);
  // f32 buffers (post_kernel rewrites several of them in place)
  float* rbuf = (float*)alloc(MC * 4);
  float* kbuf = (float*)alloc(MC * 4);  // kraw -> k_final
  float* vbuf = (float*)alloc(MC * 4);  // vraw -> v_final
  float* wbuf = (float*)alloc(MC * 4);  // w_lora -> decay
  float* abuf = (float*)alloc(MC * 4);  // a_up -> aa (=-kk)
  float* bbuf = (float*)alloc(MC * 4);  // v_up -> bb (=kk*a)
  float* gbuf = (float*)alloc(MC * 4);  // gate
  float* ybuf = (float*)alloc(MC * 4);  // wkv output

  auto tcvt = [&](const float* in, bf16_t* out, int K, int N) {
    dim3 grid(N / 32, K / 32);
    dim3 block(32, 8);
    tcvt_kernel<<<grid, block, 0, stream>>>(in, out, K, N);
  };
  tcvt(W_r, WrT, Cdim, Cdim);
  tcvt(W_k, WkT, Cdim, Cdim);
  tcvt(W_v, WvT, Cdim, Cdim);
  tcvt(W_o, WoT, Cdim, Cdim);
  tcvt(w1, w1T, Cdim, 96);   tcvt(w2, w2T, 96,  Cdim);
  tcvt(a1, a1T, Cdim, 96);   tcvt(a2, a2T, 96,  Cdim);
  tcvt(v1, v1T, Cdim, 64);   tcvt(v2, v2T, 64,  Cdim);
  tcvt(g1, g1T, Cdim, 256);  tcvt(g2, g2T, 256, Cdim);

  auto gmix = [&](const float* coef, const bf16_t* BT, int N, int mode,
                  float* outF, bf16_t* outB) {
    int jobs = (Mrows / 32) * (N / 32);
    int blocks = (jobs + 7) / 8;
    gemm_mix_kernel<<<blocks, 256, 0, stream>>>(hidden, coef, BT, N, jobs,
                                                outF, outB, mode);
  };
  auto gbf = [&](const bf16_t* A, const bf16_t* BT, int N, int K, int mode,
                 float* outF, bf16_t* outB) {
    int jobs = (Mrows / 32) * (N / 32);
    int blocks = (jobs + 7) / 8;
    gemm_bf16_kernel<<<blocks, 256, 0, stream>>>(A, BT, N, K, jobs,
                                                 outF, outB, mode);
  };

  // big projections (token-shift fused in A loader)
  gmix(x_r, WrT, Cdim, 0, rbuf, nullptr);
  gmix(x_k, WkT, Cdim, 0, kbuf, nullptr);
  gmix(x_v, WvT, Cdim, 0, vbuf, nullptr);
  // LoRA down projections (fused activations, bf16 out)
  gmix(x_w, w1T, 96,  2, nullptr, wdown);   // tanh
  gmix(x_a, a1T, 96,  1, nullptr, adown);
  gmix(x_v, v1T, 64,  1, nullptr, vdown);
  gmix(x_g, g1T, 256, 3, nullptr, gdown);   // sigmoid
  // LoRA up projections
  gbf(wdown, w2T, Cdim, 96,  0, wbuf, nullptr);
  gbf(adown, a2T, Cdim, 96,  0, abuf, nullptr);
  gbf(vdown, v2T, Cdim, 64,  0, bbuf, nullptr);
  gbf(gdown, g2T, Cdim, 256, 0, gbuf, nullptr);

  // elementwise prep (in place: wbuf->decay, kbuf->k, vbuf->v, abuf->aa, bbuf->bb)
  post_kernel<<<Mrows * Hn, Nn, 0, stream>>>(kbuf, vbuf, v_first,
                                             wbuf, abuf, bbuf,
                                             k_k, k_a, a0, w0, v0,
                                             wbuf, kbuf, vbuf, abuf, bbuf);

  // recurrent scan
  wkv_scan_kernel<<<Bsz * Hn, Nn, 0, stream>>>(rbuf, wbuf, kbuf, vbuf,
                                               abuf, bbuf, ybuf);

  // groupnorm + bonus + gate
  finalize_kernel<<<Mrows * Hn, Nn, 0, stream>>>(ybuf, rbuf, kbuf, vbuf, gbuf,
                                                 r_k, gn_w, gn_b, ypre);

  // output projection
  gbf(ypre, WoT, Cdim, Cdim, 0, (float*)d_out, nullptr);
}
